// SparseAttention_42288247996569
// MI455X (gfx1250) — compile-verified, matrix operand order checked by reading, not run on hardware
//
#include <hip/hip_runtime.h>

// ---------------- problem constants ----------------
#define H_    8
#define D_    64
#define L_    4096
#define NB    8
#define EIN   512
#define MDL_  512
#define NE_   131072
#define MROWS (NB * L_)      // 32768

typedef __attribute__((ext_vector_type(16))) __bf16 v16bf;
typedef __attribute__((ext_vector_type(8)))  __bf16 v8bf;
typedef __attribute__((ext_vector_type(8)))  float  v8f;

union FragBF { v16bf v; v8bf h[2]; };

// native fp32 -> bf16 conversion (let the backend emit v_cvt_*bf16_f32)
__device__ __forceinline__ __bf16 f2bf(float f) {
    return (__bf16)f;
}

__device__ __forceinline__ int f2ord(float f) {
    int i = __float_as_int(f);
    return (i >= 0) ? i : (i ^ 0x7fffffff);
}
__device__ __forceinline__ float ord2f(int i) {
    return __int_as_float((i >= 0) ? i : (i ^ 0x7fffffff));
}

// ------------------------------------------------------------------
// GEMM: out = X (M x K, row major) * W^T (W is Nout x K, row major) + bias
// Nout = 512 fixed. mode 0: out row-major (M x 512) ; mode 1: head-split
// store out[((n*H + h)*L + l)*64 + d]  with gm = n*L + l, gn = h*64 + d.
// Block tile 128x128, BK=32, 256 threads = 8 waves (wave32).
// Wave grid 2(m) x 4(n); each wave: 4 m-frags x 2 n-frags of 16x16.
// Double-buffered LDS with register staging: one barrier per K-step,
// next tile's global loads overlap current tile's WMMAs.
// ------------------------------------------------------------------
__global__ __launch_bounds__(256)
void gemm_bf16_wmma(const float* __restrict__ X, const float* __restrict__ W,
                    const float* __restrict__ bias, float* __restrict__ out,
                    int K, int mode)
{
    __shared__ __bf16 As[2][128][40];   // pad rows to 40 (80B, 16B-aligned chunks)
    __shared__ __bf16 Bs[2][128][40];

    const int tid  = threadIdx.x;
    const int lane = tid & 31;
    const int wid  = tid >> 5;
    const int wm   = wid >> 2;   // 0..1
    const int wn   = wid & 3;    // 0..3
    const int gm0  = blockIdx.y * 128;
    const int gn0  = blockIdx.x * 128;

    v8f acc[4][2] = {};

    const int lrow  = tid >> 3;        // 0..31
    const int lkc   = (tid & 7) * 4;   // 0,4,...,28
    const int lan15 = lane & 15;
    const int kin   = (lane >> 4) * 8; // 0 or 8 (ISA 16-bit A-matrix lane layout)

    const int nIter = K >> 5;

    // prologue: stage K-tile 0 into buffer 0
#pragma unroll
    for (int rp = 0; rp < 4; ++rp) {
        const int row = rp * 32 + lrow;
        const float4 a4 = *(const float4*)(X + (size_t)(gm0 + row) * K + lkc);
        As[0][row][lkc + 0] = f2bf(a4.x);
        As[0][row][lkc + 1] = f2bf(a4.y);
        As[0][row][lkc + 2] = f2bf(a4.z);
        As[0][row][lkc + 3] = f2bf(a4.w);
        const float4 b4 = *(const float4*)(W + (size_t)(gn0 + row) * K + lkc);
        Bs[0][row][lkc + 0] = f2bf(b4.x);
        Bs[0][row][lkc + 1] = f2bf(b4.y);
        Bs[0][row][lkc + 2] = f2bf(b4.z);
        Bs[0][row][lkc + 3] = f2bf(b4.w);
    }
    __syncthreads();

    for (int it = 0; it < nIter; ++it) {
        const int cur = it & 1;
        const int nxt = cur ^ 1;
        const bool hasNext = (it + 1) < nIter;

        // 1) issue next tile's global loads (latency hidden under WMMAs)
        float4 ra[4], rb[4];
        if (hasNext) {
            const int k0 = (it + 1) << 5;
#pragma unroll
            for (int rp = 0; rp < 4; ++rp) {
                const int row = rp * 32 + lrow;
                ra[rp] = *(const float4*)(X + (size_t)(gm0 + row) * K + k0 + lkc);
                rb[rp] = *(const float4*)(W + (size_t)(gn0 + row) * K + k0 + lkc);
            }
        }

        // 2) compute on current LDS buffer
        FragBF a[4], b[2];
#pragma unroll
        for (int mt = 0; mt < 4; ++mt) {
            const int row = wm * 64 + mt * 16 + lan15;
            a[mt].h[0] = *(const v8bf*)&As[cur][row][kin];
            a[mt].h[1] = *(const v8bf*)&As[cur][row][kin + 16];
        }
#pragma unroll
        for (int nt = 0; nt < 2; ++nt) {
            const int row = wn * 32 + nt * 16 + lan15;
            b[nt].h[0] = *(const v8bf*)&Bs[cur][row][kin];
            b[nt].h[1] = *(const v8bf*)&Bs[cur][row][kin + 16];
        }
#pragma unroll
        for (int mt = 0; mt < 4; ++mt)
#pragma unroll
            for (int nt = 0; nt < 2; ++nt)
                acc[mt][nt] = __builtin_amdgcn_wmma_f32_16x16x32_bf16(
                    false, a[mt].v, false, b[nt].v,
                    (short)0, acc[mt][nt], false, false);

        // 3) convert + store staged tile into the other buffer
        //    (safe: that buffer was last read before the previous barrier)
        if (hasNext) {
#pragma unroll
            for (int rp = 0; rp < 4; ++rp) {
                const int row = rp * 32 + lrow;
                As[nxt][row][lkc + 0] = f2bf(ra[rp].x);
                As[nxt][row][lkc + 1] = f2bf(ra[rp].y);
                As[nxt][row][lkc + 2] = f2bf(ra[rp].z);
                As[nxt][row][lkc + 3] = f2bf(ra[rp].w);
                Bs[nxt][row][lkc + 0] = f2bf(rb[rp].x);
                Bs[nxt][row][lkc + 1] = f2bf(rb[rp].y);
                Bs[nxt][row][lkc + 2] = f2bf(rb[rp].z);
                Bs[nxt][row][lkc + 3] = f2bf(rb[rp].w);
            }
        }
        __syncthreads();
    }

    // C/D layout: vgpr r -> M = r + 8*(lane>=16); N = lane&15
#pragma unroll
    for (int mt = 0; mt < 4; ++mt) {
#pragma unroll
        for (int nt = 0; nt < 2; ++nt) {
            const int gn = gn0 + wn * 32 + nt * 16 + lan15;
            const float bv = bias[gn];
#pragma unroll
            for (int r = 0; r < 8; ++r) {
                const int gm = gm0 + wm * 64 + mt * 16 + (lane >> 4) * 8 + r;
                const float val = acc[mt][nt][r] + bv;
                if (mode == 0) {
                    out[(size_t)gm * MDL_ + gn] = val;
                } else {
                    const int nb = gm >> 12;     // / L_
                    const int l  = gm & (L_ - 1);
                    const int h  = gn >> 6;      // / D_
                    const int d  = gn & (D_ - 1);
                    out[(((size_t)(nb * H_ + h)) * L_ + l) * D_ + d] = val;
                }
            }
        }
    }
}

// ------------------------------------------------------------------
// per-edge QK dot products: qk[e*64 + n*8 + h] (Q,K in (N,H,L,D) f32)
// ------------------------------------------------------------------
__global__ __launch_bounds__(256)
void edge_qk_kernel(const float* __restrict__ Qf, const float* __restrict__ Kf,
                    const int* __restrict__ adj, float* __restrict__ qk)
{
    const int idx = blockIdx.x * blockDim.x + threadIdx.x;   // < NE_*64
    const int e = idx >> 6;
    const int p = idx & 63;
    const int n = p >> 3;
    const int h = p & 7;
    const int r = adj[e];
    const int c = adj[NE_ + e];
    const float4* q4 = (const float4*)(Qf + (((size_t)(n * H_ + h)) * L_ + r) * D_);
    const float4* k4 = (const float4*)(Kf + (((size_t)(n * H_ + h)) * L_ + c) * D_);
    float s = 0.f;
#pragma unroll
    for (int i = 0; i < 16; ++i) {
        const float4 a = q4[i], b = k4[i];
        s += a.x * b.x + a.y * b.y + a.z * b.z + a.w * b.w;
    }
    qk[idx] = s * 0.125f;   // 1/sqrt(64)
}

// zero agg, init segment max/den (run AFTER edge_qk: agg aliases Q buffer)
__global__ __launch_bounds__(256)
void init_kernel(float* __restrict__ agg, int* __restrict__ mkey,
                 float* __restrict__ den)
{
    const size_t i = (size_t)blockIdx.x * blockDim.x + threadIdx.x;
    if (i < (size_t)NB * L_ * MDL_) agg[i] = 0.f;
    if (i < (size_t)L_ * 64) { mkey[i] = (int)0x80000000; den[i] = 0.f; }
}

__global__ __launch_bounds__(256)
void seg_max_kernel(const float* __restrict__ qk, const int* __restrict__ adj,
                    int* __restrict__ mkey)
{
    const int idx = blockIdx.x * blockDim.x + threadIdx.x;
    const int e = idx >> 6;
    const int p = idx & 63;
    const int r = adj[e];
    atomicMax(&mkey[r * 64 + p], f2ord(qk[idx]));
}

__global__ __launch_bounds__(256)
void seg_exp_kernel(float* __restrict__ qk, const int* __restrict__ adj,
                    const int* __restrict__ mkey, float* __restrict__ den)
{
    const int idx = blockIdx.x * blockDim.x + threadIdx.x;
    const int e = idx >> 6;
    const int p = idx & 63;
    const int r = adj[e];
    const float m = ord2f(mkey[r * 64 + p]);
    const float ex = __expf(qk[idx] - m);
    qk[idx] = ex;
    atomicAdd(&den[r * 64 + p], ex);
}

__global__ __launch_bounds__(256)
void spmm_kernel(const float* __restrict__ qk, const int* __restrict__ adj,
                 const float* __restrict__ den, const float* __restrict__ Vf,
                 float* __restrict__ agg)
{
    const int idx = blockIdx.x * blockDim.x + threadIdx.x;
    const int e = idx >> 6;
    const int p = idx & 63;
    const int n = p >> 3;
    const int h = p & 7;
    const int r = adj[e];
    const int c = adj[NE_ + e];
    const float a = qk[idx] / den[r * 64 + p];
    const float4* v4 = (const float4*)(Vf + (((size_t)(n * H_ + h)) * L_ + c) * D_);
    float* o = agg + ((size_t)n * L_ + r) * MDL_ + h * D_;
#pragma unroll
    for (int i = 0; i < 16; ++i) {
        const float4 v = v4[i];
        atomicAdd(o + i * 4 + 0, a * v.x);
        atomicAdd(o + i * 4 + 1, a * v.y);
        atomicAdd(o + i * 4 + 2, a * v.z);
        atomicAdd(o + i * 4 + 3, a * v.w);
    }
}

// ------------------------------------------------------------------
extern "C" void kernel_launch(void* const* d_in, const int* in_sizes, int n_in,
                              void* d_out, int out_size, void* d_ws, size_t ws_size,
                              hipStream_t stream)
{
    const float* queries = (const float*)d_in[0];
    const float* keys    = (const float*)d_in[1];
    const float* values  = (const float*)d_in[2];
    const int*   adj     = (const int*)d_in[3];
    const float* Wq  = (const float*)d_in[4];
    const float* bq  = (const float*)d_in[5];
    const float* Wk  = (const float*)d_in[6];
    const float* bk  = (const float*)d_in[7];
    const float* Wv  = (const float*)d_in[8];
    const float* bv  = (const float*)d_in[9];
    const float* Wfc = (const float*)d_in[10];
    const float* bfc = (const float*)d_in[11];
    float* out = (float*)d_out;

    // workspace layout (bytes)
    char* ws = (char*)d_ws;
    const size_t szQKV = (size_t)NB * H_ * L_ * D_ * sizeof(float);  // 67 MB
    const size_t szQK  = (size_t)NE_ * 64 * sizeof(float);           // 33.5 MB
    const size_t szSeg = (size_t)L_ * 64 * sizeof(float);            // 1 MB
    size_t off = 0;
    float* Qf  = (float*)(ws + off); off += szQKV;
    float* Kf  = (float*)(ws + off); off += szQKV;
    float* Vf  = (float*)(ws + off); off += szQKV;
    float* qk  = (float*)(ws + off); off += szQK;
    float* den = (float*)(ws + off); off += szSeg;
    int*   mk  = (int*)  (ws + off); off += szSeg;
    float* agg = Qf;   // Q buffer is dead after edge_qk; reuse as aggregate

    const dim3 ggrid(MDL_ / 128, MROWS / 128);   // (4, 256)
    const int  eblocks = (NE_ * 64) / 256;       // 32768
    const int  iblocks = (NB * L_ * MDL_) / 256; // 65536

    // projections (head-split output layout)
    gemm_bf16_wmma<<<ggrid, 256, 0, stream>>>(queries, Wq, bq, Qf, EIN, 1);
    gemm_bf16_wmma<<<ggrid, 256, 0, stream>>>(keys,    Wk, bk, Kf, EIN, 1);
    gemm_bf16_wmma<<<ggrid, 256, 0, stream>>>(values,  Wv, bv, Vf, EIN, 1);

    // edge QK dots (consumes Qf, Kf)
    edge_qk_kernel<<<eblocks, 256, 0, stream>>>(Qf, Kf, adj, qk);

    // init aggregate (aliases Qf) + segment state
    init_kernel<<<iblocks, 256, 0, stream>>>(agg, mk, den);

    // segment softmax
    seg_max_kernel<<<eblocks, 256, 0, stream>>>(qk, adj, mk);
    seg_exp_kernel<<<eblocks, 256, 0, stream>>>(qk, adj, mk, den);

    // weighted scatter-aggregate
    spmm_kernel<<<eblocks, 256, 0, stream>>>(qk, adj, den, Vf, agg);

    // output projection (row-major into d_out)
    gemm_bf16_wmma<<<ggrid, 256, 0, stream>>>(agg, Wfc, bfc, out, MDL_, 0);
}
